// AttentionBlock_48627619725666
// MI455X (gfx1250) — matrix operand order repaired and checked by reading, not compiled, not run on hardware
//
#include <hip/hip_runtime.h>
#include <hip/hip_bf16.h>
#include <hip/hip_fp16.h>

// ---------------------------------------------------------------------------
// AttentionBlock for MI455X (gfx1250): GroupNorm -> QKV -> MHSA -> proj (+res)
// All GEMM-shaped work runs on v_wmma_f32_16x16x32_f16 (f32 accumulate).
// GEMM wave tile: 32(M) x 64(N) = 2x4 frags -> 8 wmma per K-step, 6 loads.
// ---------------------------------------------------------------------------

typedef __attribute__((ext_vector_type(16))) _Float16 v16h;
typedef __attribute__((ext_vector_type(8)))  _Float16 h8;
typedef __attribute__((ext_vector_type(8)))  float    v8f;

#define B_      16
#define C_      512
#define N_      1024
#define NH_     8
#define HD_     64
#define GROUPS_ 32
#define EPS_    1e-5f

// D = A(16x32) * B(32x16) + C, f16 in / f32 acc
__device__ __forceinline__ v8f wmma32(v16h a, v16h b, v8f c) {
    return __builtin_amdgcn_wmma_f32_16x16x32_f16(
        /*neg_a=*/false, a, /*neg_b=*/false, b,
        /*c_mod=*/(short)0, c, /*reuse_a=*/false, /*reuse_b=*/false);
}

// A fragment (16x32, f16). Layout: lane m = l&15, half h = l>>4,
// elem i -> k = kk + 16h + i.
// Source row-major with leading dim ld (contiguous along k) -> two b128 loads.
__device__ __forceinline__ v16h load_a_frag(const _Float16* __restrict__ base,
                                            int ld, int row0, int kk, int lane) {
    int m = row0 + (lane & 15);
    int h = lane >> 4;
    const _Float16* p = base + (size_t)m * ld + kk + 16 * h;
    h8 lo = *(const h8*)(p);
    h8 hi = *(const h8*)(p + 8);
    v16h a;
#pragma unroll
    for (int i = 0; i < 8; ++i) { a[i] = lo[i]; a[i + 8] = hi[i]; }
    return a;
}

// B fragment (32x16, f16). Layout: lane n = l&15, half h = l>>4,
// elem i -> k = kk + 16h + i. Source stored n-major with leading dim ld
// (contiguous along k) -> two b128 loads.
__device__ __forceinline__ v16h load_b_frag(const _Float16* __restrict__ base,
                                            int ld, int col0, int kk, int lane) {
    int n = col0 + (lane & 15);
    int h = lane >> 4;
    const _Float16* p = base + (size_t)n * ld + kk + 16 * h;
    h8 lo = *(const h8*)(p);
    h8 hi = *(const h8*)(p + 8);
    v16h b;
#pragma unroll
    for (int i = 0; i < 8; ++i) { b[i] = lo[i]; b[i + 8] = hi[i]; }
    return b;
}

// ---------------------------------------------------------------------------
// Kernel 1: f32 -> f16 weight conversion
// ---------------------------------------------------------------------------
__global__ void __launch_bounds__(256)
k_cvt_weights(const float* __restrict__ wqkv, const float* __restrict__ wproj,
              _Float16* __restrict__ wq16, _Float16* __restrict__ wp16) {
    int i = blockIdx.x * 256 + threadIdx.x;
    if (i < 3 * C_ * C_) wq16[i] = (_Float16)wqkv[i];
    if (i < C_ * C_)     wp16[i] = (_Float16)wproj[i];
}

// ---------------------------------------------------------------------------
// Kernel 2: GroupNorm(32) + affine, store TRANSPOSED f16: Xt[B, N, C]
// One block per (batch, group): 16 channels x 1024 spatial = 16384 elems.
// ---------------------------------------------------------------------------
__global__ void __launch_bounds__(256)
k_groupnorm(const float* __restrict__ x, const float* __restrict__ gamma,
            const float* __restrict__ beta, _Float16* __restrict__ Xt) {
    int bg = blockIdx.x;             // 0..511
    int b  = bg >> 5;
    int g  = bg & 31;
    const float* xp = x + ((size_t)b * C_ + g * 16) * N_;

    __shared__ float ssum[256];
    __shared__ float ssq[256];
    float s = 0.f, q = 0.f;
    for (int i = threadIdx.x; i < 16 * N_; i += 256) {
        float v = xp[i];
        s += v; q += v * v;
    }
    ssum[threadIdx.x] = s; ssq[threadIdx.x] = q;
    __syncthreads();
    for (int off = 128; off > 0; off >>= 1) {
        if (threadIdx.x < off) {
            ssum[threadIdx.x] += ssum[threadIdx.x + off];
            ssq[threadIdx.x]  += ssq[threadIdx.x + off];
        }
        __syncthreads();
    }
    const float inv_n = 1.f / (16.f * (float)N_);
    float mean = ssum[0] * inv_n;
    float var  = ssq[0] * inv_n - mean * mean;
    float rstd = rsqrtf(var + EPS_);

    for (int i = threadIdx.x; i < 16 * N_; i += 256) {
        int ch = i >> 10;            // 0..15
        int n  = i & (N_ - 1);
        int c  = g * 16 + ch;
        float v = (xp[i] - mean) * rstd * gamma[c] + beta[c];
        Xt[((size_t)b * N_ + n) * C_ + c] = (_Float16)v;
    }
}

// ---------------------------------------------------------------------------
// Kernel 3: QKV GEMM per batch: out[o,n] = Wqkv[o,:] . Xt[b,n,:] + bias[o]
// Block tile 64(M) x 256(N), 8 waves, each wave 32x64 (2x4 wmma frags).
// Epilogue scatters into Q,K = [B,h,N,64] (b128 stores), Vt = [B,h,64,N].
// Scale 1/sqrt(hd) folded into Q.
// ---------------------------------------------------------------------------
__device__ __forceinline__ void store_qkv_frag(v8f c, int b, int o0, int n0,
                                               int lane,
                                               const float* __restrict__ bias,
                                               _Float16* __restrict__ Q,
                                               _Float16* __restrict__ K,
                                               _Float16* __restrict__ Vt) {
    int n = n0 + (lane & 15);
    int h = lane >> 4;
    if (o0 < C_) {                                   // ---- Q (scaled)
        int head = o0 >> 6;
        int d0   = (o0 & 63) + 8 * h;
        h8 v;
#pragma unroll
        for (int i = 0; i < 8; ++i) {
            int o = o0 + i + 8 * h;
            v[i] = (_Float16)((c[i] + bias[o]) * 0.125f);  // hd^-0.5 = 1/8
        }
        *(h8*)(Q + (((size_t)b * NH_ + head) * N_ + n) * HD_ + d0) = v;
    } else if (o0 < 2 * C_) {                        // ---- K
        int oc   = o0 - C_;
        int head = oc >> 6;
        int d0   = (oc & 63) + 8 * h;
        h8 v;
#pragma unroll
        for (int i = 0; i < 8; ++i) {
            int o = o0 + i + 8 * h;
            v[i] = (_Float16)(c[i] + bias[o]);
        }
        *(h8*)(K + (((size_t)b * NH_ + head) * N_ + n) * HD_ + d0) = v;
    } else {                                         // ---- V (transposed)
        int oc   = o0 - 2 * C_;
        int head = oc >> 6;
#pragma unroll
        for (int i = 0; i < 8; ++i) {
            int o = o0 + i + 8 * h;
            int d = (oc & 63) + i + 8 * h;
            Vt[(((size_t)b * NH_ + head) * HD_ + d) * N_ + n] =
                (_Float16)(c[i] + bias[o]);
        }
    }
}

__global__ void __launch_bounds__(256)
k_qkv_gemm(const _Float16* __restrict__ W,      // [1536, 512]
           const float*    __restrict__ bias,   // [1536]
           const _Float16* __restrict__ Xt,     // [B, N, C]
           _Float16* __restrict__ Q, _Float16* __restrict__ K,
           _Float16* __restrict__ Vt) {
    int lane = threadIdx.x & 31;
    int wv   = threadIdx.x >> 5;
    int b    = blockIdx.z;
    int m0   = blockIdx.y * 64 + (wv >> 2) * 32;    // 2 wave-rows of 32
    int n0   = blockIdx.x * 256 + (wv & 3) * 64;    // 4 wave-cols of 64
    const _Float16* Xb = Xt + (size_t)b * N_ * C_;

    v8f acc[2][4];
#pragma unroll
    for (int r = 0; r < 2; ++r)
#pragma unroll
        for (int s = 0; s < 4; ++s) acc[r][s] = (v8f){};

#pragma unroll 2
    for (int kk = 0; kk < C_; kk += 32) {
        v16h a0 = load_a_frag(W, C_, m0,      kk, lane);
        v16h a1 = load_a_frag(W, C_, m0 + 16, kk, lane);
#pragma unroll
        for (int s = 0; s < 4; ++s) {
            v16h bf = load_b_frag(Xb, C_, n0 + 16 * s, kk, lane);
            acc[0][s] = wmma32(a0, bf, acc[0][s]);
            acc[1][s] = wmma32(a1, bf, acc[1][s]);
        }
    }
#pragma unroll
    for (int r = 0; r < 2; ++r)
#pragma unroll
        for (int s = 0; s < 4; ++s)
            store_qkv_frag(acc[r][s], b, m0 + 16 * r, n0 + 16 * s,
                           lane, bias, Q, K, Vt);
}

// ---------------------------------------------------------------------------
// Kernel 4: flash attention, transposed formulation.
// One wave per 16-query tile: S^T(16k x 16q) = K_tile(16x64) @ Q^T(64x16),
// online softmax is lane-local over accumulator elements (keys on elem axis),
// O^T(64d x 16q) += V^T @ P^T with P^T -> B-frag via per-elem cvt + shfl_xor16.
// Output written as rows of Ht[B, N, C] (b128 per frag per lane).
// ---------------------------------------------------------------------------
__global__ void __launch_bounds__(256)
k_attention(const _Float16* __restrict__ Q, const _Float16* __restrict__ K,
            const _Float16* __restrict__ Vt, _Float16* __restrict__ Ht) {
    int lane = threadIdx.x & 31;
    int wv   = threadIdx.x >> 5;
    int gw   = blockIdx.x * 8 + wv;       // 8192 waves total
    int qt   = gw & 63;                   // query tile 0..63
    int bh   = gw >> 6;                   // (batch*heads) 0..127
    int b    = bh >> 3;
    int head = bh & 7;
    int q0   = qt * 16;
    int h    = lane >> 4;

    const _Float16* Qb = Q  + (size_t)bh * N_ * HD_;
    const _Float16* Kb = K  + (size_t)bh * N_ * HD_;
    const _Float16* Vb = Vt + (size_t)bh * HD_ * N_;

    // Q^T B-fragments (reused for all key blocks): lane n=query, k=channel.
    v16h qb0 = load_b_frag(Qb, HD_, q0, 0,  lane);
    v16h qb1 = load_b_frag(Qb, HD_, q0, 32, lane);

    v8f o0 = {}, o1 = {}, o2 = {}, o3 = {};
    float run_max = -3.4e38f;
    float run_sum = 0.f;

    for (int k0 = 0; k0 < N_; k0 += 32) {
        // S^T for keys [k0, k0+16) and [k0+16, k0+32)
        v16h kA0 = load_a_frag(Kb, HD_, k0,      0,  lane);
        v16h kA1 = load_a_frag(Kb, HD_, k0,      32, lane);
        v16h kB0 = load_a_frag(Kb, HD_, k0 + 16, 0,  lane);
        v16h kB1 = load_a_frag(Kb, HD_, k0 + 16, 32, lane);
        v8f s0 = {}, s1 = {};
        s0 = wmma32(kA0, qb0, s0);
        s0 = wmma32(kA1, qb1, s0);
        s1 = wmma32(kB0, qb0, s1);
        s1 = wmma32(kB1, qb1, s1);

        // ---- online softmax (keys live on elem axis -> lane-local) ----
        float mloc = -3.4e38f;
#pragma unroll
        for (int i = 0; i < 8; ++i) mloc = fmaxf(mloc, fmaxf(s0[i], s1[i]));
        float mblk = fmaxf(mloc, __shfl_xor(mloc, 16, 32)); // combine lane pair
        float mnew = fmaxf(run_max, mblk);
        float corr = __expf(run_max - mnew);

        float p0[8], p1[8];
        float lsum = 0.f;
#pragma unroll
        for (int i = 0; i < 8; ++i) {
            p0[i] = __expf(s0[i] - mnew);
            p1[i] = __expf(s1[i] - mnew);
            lsum += p0[i] + p1[i];
        }
        run_sum = run_sum * corr + (lsum + __shfl_xor(lsum, 16, 32));
        run_max = mnew;

#pragma unroll
        for (int i = 0; i < 8; ++i) {
            o0[i] *= corr; o1[i] *= corr; o2[i] *= corr; o3[i] *= corr;
        }

        // ---- P^T (C-layout f32) -> B-fragment (f16), half-wave swap ----
        v16h pb;
#pragma unroll
        for (int j = 0; j < 8; ++j) {
            float sx0 = __shfl_xor(p0[j], 16, 32);
            float sx1 = __shfl_xor(p1[j], 16, 32);
            pb[j]     = (_Float16)((h == 0) ? p0[j] : sx1);
            pb[8 + j] = (_Float16)((h == 0) ? sx0 : p1[j]);
        }

        // ---- O^T += V^T(64 x 32keys) @ P^T(32keys x 16q) ----
        v16h vA0 = load_a_frag(Vb, N_, 0,  k0, lane);
        v16h vA1 = load_a_frag(Vb, N_, 16, k0, lane);
        v16h vA2 = load_a_frag(Vb, N_, 32, k0, lane);
        v16h vA3 = load_a_frag(Vb, N_, 48, k0, lane);
        o0 = wmma32(vA0, pb, o0);
        o1 = wmma32(vA1, pb, o1);
        o2 = wmma32(vA2, pb, o2);
        o3 = wmma32(vA3, pb, o3);
    }

    float inv = 1.f / run_sum;
    int q = q0 + (lane & 15);
    _Float16* dst = Ht + ((size_t)b * N_ + q) * C_ + head * HD_;
    v8f fr[4] = { o0, o1, o2, o3 };
#pragma unroll
    for (int t = 0; t < 4; ++t) {
        h8 v;
#pragma unroll
        for (int i = 0; i < 8; ++i) v[i] = (_Float16)(fr[t][i] * inv);
        *(h8*)(dst + t * 16 + 8 * h) = v;   // d = 16t + i + 8h contiguous
    }
}

// ---------------------------------------------------------------------------
// Kernel 5: output projection + bias + residual, f32 out [B, C, N]
// Block tile 64(M) x 256(N), wave tile 32x64 (2x4 frags).
// ---------------------------------------------------------------------------
__global__ void __launch_bounds__(256)
k_proj_gemm(const _Float16* __restrict__ Wp,    // [512, 512] f16
            const float*    __restrict__ bias,  // [512]
            const _Float16* __restrict__ Ht,    // [B, N, C] f16
            const float*    __restrict__ x,     // residual [B, C, N]
            float* __restrict__ out) {
    int lane = threadIdx.x & 31;
    int wv   = threadIdx.x >> 5;
    int b    = blockIdx.z;
    int m0   = blockIdx.y * 64 + (wv >> 2) * 32;
    int n0   = blockIdx.x * 256 + (wv & 3) * 64;
    const _Float16* Hb = Ht + (size_t)b * N_ * C_;

    v8f acc[2][4];
#pragma unroll
    for (int r = 0; r < 2; ++r)
#pragma unroll
        for (int s = 0; s < 4; ++s) acc[r][s] = (v8f){};

#pragma unroll 2
    for (int kk = 0; kk < C_; kk += 32) {
        v16h a0 = load_a_frag(Wp, C_, m0,      kk, lane);
        v16h a1 = load_a_frag(Wp, C_, m0 + 16, kk, lane);
#pragma unroll
        for (int s = 0; s < 4; ++s) {
            v16h bf = load_b_frag(Hb, C_, n0 + 16 * s, kk, lane);
            acc[0][s] = wmma32(a0, bf, acc[0][s]);
            acc[1][s] = wmma32(a1, bf, acc[1][s]);
        }
    }

    int h = lane >> 4;
#pragma unroll
    for (int r = 0; r < 2; ++r) {
#pragma unroll
        for (int s = 0; s < 4; ++s) {
            int n = n0 + 16 * s + (lane & 15);
#pragma unroll
            for (int i = 0; i < 8; ++i) {
                int o = m0 + 16 * r + i + 8 * h;
                size_t idx = ((size_t)b * C_ + o) * N_ + n;
                out[idx] = acc[r][s][i] + bias[o] + x[idx];
            }
        }
    }
}

// ---------------------------------------------------------------------------
// Launch
// ---------------------------------------------------------------------------
extern "C" void kernel_launch(void* const* d_in, const int* in_sizes, int n_in,
                              void* d_out, int out_size, void* d_ws, size_t ws_size,
                              hipStream_t stream) {
    const float* x      = (const float*)d_in[0];
    const float* gamma  = (const float*)d_in[1];
    const float* beta   = (const float*)d_in[2];
    const float* w_qkv  = (const float*)d_in[3];
    const float* b_qkv  = (const float*)d_in[4];
    const float* w_proj = (const float*)d_in[5];
    const float* b_proj = (const float*)d_in[6];
    float* out = (float*)d_out;

    // Workspace layout (bytes), all 256B-aligned. Ht aliases Xt (stream-ordered
    // reuse: QKV consumes Xt before attention writes Ht).
    char* ws = (char*)d_ws;
    _Float16* Xt  = (_Float16*)(ws + 0);          // [B,N,C]  16 MB (also Ht)
    _Float16* Wq  = (_Float16*)(ws + 16777216);   // [1536,512] 1.5 MB
    _Float16* Wp  = (_Float16*)(ws + 18350080);   // [512,512]  0.5 MB
    _Float16* Qd  = (_Float16*)(ws + 18874368);   // [B,h,N,64] 16 MB
    _Float16* Kd  = (_Float16*)(ws + 35651584);   // [B,h,N,64] 16 MB
    _Float16* Vt  = (_Float16*)(ws + 52428800);   // [B,h,64,N] 16 MB
    _Float16* Ht  = Xt;

    k_cvt_weights<<<(3 * C_ * C_) / 256, 256, 0, stream>>>(w_qkv, w_proj, Wq, Wp);
    k_groupnorm<<<B_ * GROUPS_, 256, 0, stream>>>(x, gamma, beta, Xt);
    k_qkv_gemm<<<dim3(N_ / 256, (3 * C_) / 64, B_), 256, 0, stream>>>(
        Wq, b_qkv, Xt, Qd, Kd, Vt);
    k_attention<<<(B_ * NH_ * (N_ / 16)) / 8, 256, 0, stream>>>(Qd, Kd, Vt, Ht);
    k_proj_gemm<<<dim3(N_ / 256, C_ / 64, B_), 256, 0, stream>>>(
        Wp, b_proj, Ht, x, out);
}
